// Body2_interaction_15367392985803
// MI455X (gfx1250) — matrix operand order repaired and checked by reading, not compiled, not run on hardware
//
#include <hip/hip_runtime.h>
#include <hip/hip_bf16.h>

// Fused SO(3) block: out = Lin_final( TP_uuu( Lin_left(x), Lin_right(x) ) )
//
// Roofline (MI455X): 44.2 GFLOP of 128-K GEMMs vs 460 MB HBM traffic
// (x fp32 in + out fp32; weights live in the 192MB L2). ~96 FLOP/B =>
// run the GEMMs on the WMMA pipe (v_wmma_f32_16x16x32_f16, f32 accum)
// and fuse all three linears + the CG tensor product in one kernel so
// the [N,9,128] intermediates never leave LDS/registers.
//
// Per 256-thread block (8 wave32): 16 samples. LDS tile [m*16+s][128] f16,
// row padded by 8 halves (stride 136) to kill bank conflicts on the
// 256B-stride A-fragment loads. Each wave owns a fixed 16-column slice;
// the f32 C/D WMMA layout puts the channel in the lane index, so the
// sparse Wigner-3j contraction is pure in-lane VALU with hard-coded
// coefficients. l=0 rows of xl/xr (and b_left/b_right) are dead in the
// reference TP and are skipped.
//
// v2: batch all 4 A-fragments of a tile before the 8-WMMA chain so the
// ds_load_b128s issue as one clause instead of load/wait/wmma lockstep.

typedef __attribute__((ext_vector_type(16))) _Float16 v16h;
typedef __attribute__((ext_vector_type(8)))  _Float16 v8h;
typedef __attribute__((ext_vector_type(4)))  _Float16 v4h;
typedef __attribute__((ext_vector_type(8)))  float    v8f;

#define LDS_ROW 136   // 128 + 8 halves pad (272 B, 16B-aligned, bank skew 4)

static __device__ __forceinline__ v8f wmma_f16(v16h a, v16h b, v8f c) {
  // (neg_a, A, neg_b, B, c_mod, C, reuse_a, reuse_b)
  return __builtin_amdgcn_wmma_f32_16x16x32_f16(false, a, false, b,
                                                (short)0, c, false, false);
}

// A fragment: 16x32 f16 tile, rows = 16 samples of one m, from LDS.
// ISA layout: lane<16 -> K=[0..7]+[16..23], lane>=16 -> K=[8..15]+[24..31].
static __device__ __forceinline__ v16h a_frag(const _Float16* sm, int mt,
                                              int lane, int kt) {
  const _Float16* row = sm + (mt * 16 + (lane & 15)) * LDS_ROW;
  const int off = ((lane >> 4) << 3) + (kt << 5);
  v8h lo = *reinterpret_cast<const v8h*>(row + off);
  v8h hi = *reinterpret_cast<const v8h*>(row + off + 16);
  v16h a;
#pragma unroll
  for (int q = 0; q < 8; ++q) { a[q] = lo[q]; a[q + 8] = hi[q]; }
  return a;
}

// B fragment: B[k][n] = W^T[k][n] = W[n][k]; per lane read 16 (two runs of 8, 16 apart)
// fp32 of W row (col) starting at K-offset, convert to f16 inline.
// ISA layout: lane<16 -> K=[0..7]+[16..23], lane>=16 -> K=[8..15]+[24..31].
static __device__ __forceinline__ v16h b_frag(const float* __restrict__ W,
                                              int l, int col, int lane, int kt) {
  const int ks = (kt << 5) + ((lane >> 4) << 3);
  const float* p = W + ((size_t)(l * 128 + col) * 128 + ks);
  v16h b;
#pragma unroll
  for (int q = 0; q < 16; ++q) b[q] = (_Float16)p[q < 8 ? q : q + 8];
  return b;
}

__global__ __launch_bounds__(256) void so3_fused_kernel(
    const float* __restrict__ x,    // [N,9,128]
    const float* __restrict__ Wl,   // [3,128,128]
    const float* __restrict__ Wr,   // [3,128,128]
    const float* __restrict__ Wf,   // [3,128,128]
    const float* __restrict__ bf,   // [128]
    float* __restrict__ out,        // [N,9,128]
    int nsamp) {
  __shared__ _Float16 smem[144 * LDS_ROW];  // 38.25 KB

  const int tid   = threadIdx.x;
  const int lane  = tid & 31;
  const int wave  = tid >> 5;
  const int col   = (wave << 4) + (lane & 15);  // this lane's channel/column
  const int sbase = (lane >> 4) << 3;           // sample base in D layout
  const int s0    = blockIdx.x << 4;            // first sample of block

  // ---- stage 0: x tile fp32 -> f16 LDS, layout [m*16+s][c] ----
  {
    const float* xg = x + (size_t)s0 * 9 * 128;
#pragma unroll
    for (int it = 0; it < 18; ++it) {            // 4608 float4 / 256 thr
      const int i  = tid + (it << 8);
      const int r  = i >> 5;                     // (s*9+m), 0..143
      const int c4 = (i & 31) << 2;
      const int s  = r / 9;
      const int m  = r - s * 9;
      float4 f = make_float4(0.f, 0.f, 0.f, 0.f);
      if (s0 + s < nsamp)
        f = *reinterpret_cast<const float4*>(xg + (size_t)r * 128 + c4);
      v4h h = {(_Float16)f.x, (_Float16)f.y, (_Float16)f.z, (_Float16)f.w};
      *reinterpret_cast<v4h*>(&smem[(m * 16 + s) * LDS_ROW + c4]) = h;
    }
  }
  __syncthreads();

  // ---- stage A: xl = x @ Wl^T, xr = x @ Wr^T for m=1..8 (l=0 is dead) ----
  v8f accL[8], accR[8];
#pragma unroll
  for (int t = 0; t < 8; ++t) {
    v8f z = {0.f, 0.f, 0.f, 0.f, 0.f, 0.f, 0.f, 0.f};
    accL[t] = z; accR[t] = z;
  }
#pragma unroll
  for (int l = 1; l <= 2; ++l) {
    v16h BL[4], BR[4];
#pragma unroll
    for (int kt = 0; kt < 4; ++kt) {
      BL[kt] = b_frag(Wl, l, col, lane, kt);
      BR[kt] = b_frag(Wr, l, col, lane, kt);
    }
#pragma unroll
    for (int m = l * l; m < (l + 1) * (l + 1); ++m) {
      v16h a[4];
#pragma unroll
      for (int kt = 0; kt < 4; ++kt) a[kt] = a_frag(smem, m, lane, kt);
#pragma unroll
      for (int kt = 0; kt < 4; ++kt) {
        accL[m - 1] = wmma_f16(a[kt], BL[kt], accL[m - 1]);
        accR[m - 1] = wmma_f16(a[kt], BR[kt], accR[m - 1]);
      }
    }
  }

  // ---- stage B: sparse CG tensor product, in-lane (channel = lane) ----
  // acc*[t][v]: m = t+1, sample = sbase+v, channel = col.
  // Coefficients = e3nn unit-norm wigner_3j scaled by alpha=sqrt(2*lo+1).
  float tout[9][8];
#pragma unroll
  for (int v = 0; v < 8; ++v) {
    const float xl0 = accL[0][v], xl1 = accL[1][v], xl2 = accL[2][v];
    const float xl3 = accL[3][v], xl4 = accL[4][v], xl5 = accL[5][v];
    const float xl6 = accL[6][v], xl7 = accL[7][v];
    const float xr0 = accR[0][v], xr1 = accR[1][v], xr2 = accR[2][v];
    const float xr3 = accR[3][v], xr4 = accR[4][v], xr5 = accR[5][v];
    const float xr6 = accR[6][v], xr7 = accR[7][v];
    // (2,2,0): identity/sqrt(5), alpha=1
    tout[0][v] = 0.44721360f *
        (xl3 * xr3 + xl4 * xr4 + xl5 * xr5 + xl6 * xr6 + xl7 * xr7);
    // (1,2,1): alpha=sqrt(3)
    tout[1][v] = 0.54772256f * xl2 * xr3 + 0.54772256f * xl1 * xr4
               - 0.31622777f * xl0 * xr5 - 0.54772256f * xl0 * xr7;
    tout[2][v] = 0.54772256f * xl0 * xr4 + 0.63245553f * xl1 * xr5
               + 0.54772256f * xl2 * xr6;
    tout[3][v] = 0.54772256f * xl0 * xr3 - 0.31622777f * xl2 * xr5
               + 0.54772256f * xl1 * xr6 + 0.54772256f * xl2 * xr7;
    // (1,1,2): alpha=sqrt(5)
    tout[4][v] = 0.70710678f * (xl2 * xr0 + xl0 * xr2);
    tout[5][v] = 0.70710678f * (xl0 * xr1 + xl1 * xr0);
    tout[6][v] = 0.81649658f * xl1 * xr1
               - 0.40824829f * (xl0 * xr0 + xl2 * xr2);
    tout[7][v] = 0.70710678f * (xl2 * xr1 + xl1 * xr2);
    tout[8][v] = 0.70710678f * xl2 * xr2 - 0.70710678f * xl0 * xr0;
  }

  __syncthreads();  // all waves done reading x-tile A fragments
#pragma unroll
  for (int m = 0; m < 9; ++m)
#pragma unroll
    for (int v = 0; v < 8; ++v)
      smem[(m * 16 + sbase + v) * LDS_ROW + col] = (_Float16)tout[m][v];
  __syncthreads();

  // ---- stage C: out = tp @ Wf^T (+ bf on l=0 rows) ----
  const float bias0 = bf[col];
#pragma unroll
  for (int l = 0; l < 3; ++l) {
    v16h BF[4];
#pragma unroll
    for (int kt = 0; kt < 4; ++kt) BF[kt] = b_frag(Wf, l, col, lane, kt);
#pragma unroll
    for (int m = l * l; m < (l + 1) * (l + 1); ++m) {
      v16h a[4];
#pragma unroll
      for (int kt = 0; kt < 4; ++kt) a[kt] = a_frag(smem, m, lane, kt);
      v8f acc = {0.f, 0.f, 0.f, 0.f, 0.f, 0.f, 0.f, 0.f};
#pragma unroll
      for (int kt = 0; kt < 4; ++kt)
        acc = wmma_f16(a[kt], BF[kt], acc);
      if (l == 0) {
#pragma unroll
        for (int v = 0; v < 8; ++v) acc[v] += bias0;
      }
#pragma unroll
      for (int v = 0; v < 8; ++v) {
        const int gs = s0 + sbase + v;
        if (gs < nsamp)
          out[((size_t)gs * 9 + m) * 128 + col] = acc[v];
      }
    }
  }
}

extern "C" void kernel_launch(void* const* d_in, const int* in_sizes, int n_in,
                              void* d_out, int out_size, void* d_ws, size_t ws_size,
                              hipStream_t stream) {
  const float* x  = (const float*)d_in[0];  // irreps_x [N,9,128]
  const float* Wl = (const float*)d_in[1];  // W_left  [3,128,128]
  // d_in[2] = b_left  : dead (l=0 inputs unused by the tensor product)
  const float* Wr = (const float*)d_in[3];  // W_right [3,128,128]
  // d_in[4] = b_right : dead
  const float* Wf = (const float*)d_in[5];  // W_final [3,128,128]
  const float* bf = (const float*)d_in[6];  // b_final [128]
  float* out = (float*)d_out;

  const int nsamp  = in_sizes[0] / (9 * 128);
  const int blocks = (nsamp + 15) / 16;
  so3_fused_kernel<<<blocks, 256, 0, stream>>>(x, Wl, Wr, Wf, bf, out, nsamp);
}
